// Transformer_62208306315486
// MI455X (gfx1250) — matrix operand order repaired and checked by reading, not compiled, not run on hardware
//
#include <hip/hip_runtime.h>
#include <hip/hip_bf16.h>

// ---------------------------------------------------------------------------
// Transformer encoder-decoder forward on MI455X (gfx1250).
// GEMMs: V_WMMA_F32_16X16X32_F16, f16 in / f32 accumulate.
// Tiles staged to LDS with GLOBAL_LOAD_ASYNC_TO_LDS_B128 (ASYNCcnt),
// double-buffered so async HBM->LDS overlaps the WMMA pipe.
// ---------------------------------------------------------------------------

typedef _Float16 h16;
typedef __attribute__((ext_vector_type(16))) _Float16 v16h;
typedef __attribute__((ext_vector_type(8)))  float    v8f;

#define TPB 256
#define LDK 40          // padded LDS row stride (h16): 32 + 8 keeps 16B align

// ---- CDNA5 async copy: global -> LDS, 16B per lane, tracked by ASYNCcnt ----
__device__ __forceinline__ unsigned lds_off32(const void* p) {
    // flat address of __shared__ = {SHARED_BASE, lds_offset}; low 32 bits are
    // the LDS byte address the VDST VGPR of async-load-to-LDS expects.
    return (unsigned)(unsigned long long)p;
}
__device__ __forceinline__ void async_cp_b128(const void* lds_dst, const void* gsrc) {
    asm volatile("global_load_async_to_lds_b128 %0, %1, off"
                 :: "v"(lds_off32(lds_dst)), "v"(gsrc) : "memory");
}
__device__ __forceinline__ void wait_async0() {
    asm volatile("s_wait_asynccnt 0x0" ::: "memory");
}

// ------------------------- block reductions --------------------------------
__device__ __forceinline__ float blk_sum(float v) {
    __shared__ float red[TPB];
    const int t = threadIdx.x;
    red[t] = v; __syncthreads();
    #pragma unroll
    for (int s = TPB / 2; s > 0; s >>= 1) {
        if (t < s) red[t] += red[t + s];
        __syncthreads();
    }
    float r = red[0]; __syncthreads();
    return r;
}

__device__ __forceinline__ float blk_max(float v) {
    __shared__ float red[TPB];
    const int t = threadIdx.x;
    red[t] = v; __syncthreads();
    #pragma unroll
    for (int s = TPB / 2; s > 0; s >>= 1) {
        if (t < s) red[t] = fmaxf(red[t], red[t + s]);
        __syncthreads();
    }
    float r = red[0]; __syncthreads();
    return r;
}

// ------------------------- f32 -> f16 convert ------------------------------
__global__ __launch_bounds__(TPB) void cvt_k(const float* __restrict__ s,
                                             h16* __restrict__ d, long n) {
    long i = (long)blockIdx.x * TPB + threadIdx.x;
    const long stride = (long)gridDim.x * TPB;
    for (; i < n; i += stride) d[i] = (h16)s[i];
}

// ---- transpose-convert: src [K,N] f32 (layer z) -> dst [N,K] f16 ----------
__global__ __launch_bounds__(TPB) void cvt_t_k(const float* __restrict__ s,
                                               h16* __restrict__ d, int K, int N) {
    __shared__ float tile[32][33];
    const long base = (long)blockIdx.z * (long)K * N;
    const int k0 = blockIdx.y * 32;
    const int n0 = blockIdx.x * 32;
    const int tx = threadIdx.x & 31;
    const int ty = threadIdx.x >> 5;            // 0..7
    #pragma unroll
    for (int i = 0; i < 4; ++i)
        tile[ty + 8 * i][tx] = s[base + (long)(k0 + ty + 8 * i) * N + n0 + tx];
    __syncthreads();
    #pragma unroll
    for (int i = 0; i < 4; ++i)
        d[base + (long)(n0 + ty + 8 * i) * K + k0 + tx] = (h16)tile[tx][ty + 8 * i];
}

// ------------------------- WMMA GEMM ---------------------------------------
// C[M,N] = alpha * A[M,K] @ (BT ? Bsrc[N,K]^T : Bsrc[K,N])
// Block tile 128x128, K-step 32, double-buffered LDS, 8 waves (4M x 2N),
// each wave a 32x64 tile = 2x4 WMMA fragments.
// M%128==0, N%128==0, K%32==0 by construction of the network shapes.
template<int BT, int OUTH>
__global__ __launch_bounds__(TPB) void gemm_h16(
    const h16* __restrict__ A, const h16* __restrict__ Bm,
    float* __restrict__ C, h16* __restrict__ Ch,
    int M, int N, int K, long sA, long sB, long sC, float alpha)
{
    __shared__ __align__(16) h16 Al[2][128][LDK];
    __shared__ __align__(16) h16 Bl[2][128][LDK];

    const long bz = blockIdx.z;
    A  += bz * sA;
    Bm += bz * sB;
    if (OUTH) Ch += bz * sC; else C += bz * sC;

    const int tid  = threadIdx.x;
    const int lane = tid & 31;
    const int wave = tid >> 5;
    const int wm   = (wave >> 1) * 32;          // 4 waves along M
    const int wn   = (wave & 1) * 64;           // 2 waves along N
    const int row0 = blockIdx.y * 128;
    const int col0 = blockIdx.x * 128;
    const int m16  = lane & 15;
    const int hi   = lane >> 4;

    const int ar = tid >> 2;                    // 0..63 (A / B-T rows)
    const int ac = (tid & 3) * 8;               // 0,8,16,24 (k offset)
    const int bk = tid >> 4;                    // 0..15 (B-NN k row)
    const int bn = (tid & 15) * 8;              // 0..120 (B-NN n offset)

    v8f acc[2][4] = {};

    auto load_tiles = [&](int k0, int buf) {
        #pragma unroll
        for (int rep = 0; rep < 2; ++rep) {
            async_cp_b128(&Al[buf][ar + rep * 64][ac],
                          A + (long)(row0 + ar + rep * 64) * K + (k0 + ac));
        }
        if (BT) {
            #pragma unroll
            for (int rep = 0; rep < 2; ++rep) {
                async_cp_b128(&Bl[buf][ar + rep * 64][ac],
                              Bm + (long)(col0 + ar + rep * 64) * K + (k0 + ac));
            }
        } else {
            // Bsrc [K,N]: coalesced 8xf16 read along n, transposed LDS write
            #pragma unroll
            for (int rep = 0; rep < 2; ++rep) {
                union { uint4 u; h16 h[8]; } t;
                t.u = *(const uint4*)(Bm + (long)(k0 + bk + rep * 16) * N + (col0 + bn));
                #pragma unroll
                for (int i = 0; i < 8; ++i) Bl[buf][bn + i][bk + rep * 16] = t.h[i];
            }
        }
    };

    load_tiles(0, 0);
    int buf = 0;
    for (int k0 = 0; k0 < K; k0 += 32) {
        wait_async0();
        __syncthreads();
        if (k0 + 32 < K) load_tiles(k0 + 32, buf ^ 1);

        // ---- fragments (CDNA5 documented VGPR layouts) ----
        union { v16h v; unsigned u[8]; } af[2], bf[4];
        #pragma unroll
        for (int r = 0; r < 2; ++r) {
            const h16* arow = &Al[buf][wm + r * 16 + m16][0];
            #pragma unroll
            for (int j = 0; j < 8; ++j) {
                // lanes 0-15: K {0..7,16..23}; lanes 16-31: K {8..15,24..31}
                const int kk = ((j & 3) * 2) + ((j >> 2) * 16) + hi * 8;
                af[r].u[j] = *(const unsigned*)(arow + kk);
            }
        }
        #pragma unroll
        for (int c = 0; c < 4; ++c) {
            const h16* brow = &Bl[buf][wn + c * 16 + m16][0];
            #pragma unroll
            for (int j = 0; j < 8; ++j) {
                // lanes 0-15: K {0..7,16..23}; lanes 16-31: K {8..15,24..31}
                const int kk = ((j & 3) * 2) + ((j >> 2) * 16) + hi * 8;
                bf[c].u[j] = *(const unsigned*)(brow + kk);
            }
        }
        #pragma unroll
        for (int r = 0; r < 2; ++r)
            #pragma unroll
            for (int c = 0; c < 4; ++c)
                acc[r][c] = __builtin_amdgcn_wmma_f32_16x16x32_f16(
                    false, af[r].v, false, bf[c].v, (short)0, acc[r][c], false, false);
        buf ^= 1;
    }

    // ---- epilogue: C/D layout: VGPR i -> row (hi*8 + i), lane m16 -> col ----
    #pragma unroll
    for (int r = 0; r < 2; ++r) {
        #pragma unroll
        for (int c = 0; c < 4; ++c) {
            const int col = col0 + wn + c * 16 + m16;
            const int rb  = row0 + wm + r * 16 + hi * 8;
            #pragma unroll
            for (int i = 0; i < 8; ++i) {
                const float v = acc[r][c][i] * alpha;
                if (OUTH) Ch[(long)(rb + i) * N + col] = (h16)v;
                else      C [(long)(rb + i) * N + col] = v;
            }
        }
    }
}

// ------------------------- softmax (2048 cols -> f16 probs) ----------------
__global__ __launch_bounds__(TPB) void softmax2048_k(const float* __restrict__ S,
                                                     h16* __restrict__ P) {
    const long row = blockIdx.x;
    const float* s = S + row * 2048;
    h16* p = P + row * 2048;
    const int t = threadIdx.x;
    float loc[8];
    float m = -3.0e38f;
    #pragma unroll
    for (int i = 0; i < 8; ++i) { loc[i] = s[t + i * TPB]; m = fmaxf(m, loc[i]); }
    m = blk_max(m);
    float sum = 0.f;
    #pragma unroll
    for (int i = 0; i < 8; ++i) { loc[i] = __expf(loc[i] - m); sum += loc[i]; }
    sum = blk_sum(sum);
    const float inv = 1.0f / sum;
    #pragma unroll
    for (int i = 0; i < 8; ++i) p[t + i * TPB] = (h16)(loc[i] * inv);
}

// ------------------------- residual + layernorm (opt. affine) --------------
__global__ __launch_bounds__(TPB) void add_ln_k(const float* __restrict__ T,
                                                const float* __restrict__ R,
                                                const float* __restrict__ g,
                                                const float* __restrict__ b,
                                                float* __restrict__ Of,
                                                h16* __restrict__ Oh) {
    const long o = (long)blockIdx.x * 512;
    const int t = threadIdx.x;
    float z0 = T[o + t]       + R[o + t];
    float z1 = T[o + t + 256] + R[o + t + 256];
    const float mean = blk_sum(z0 + z1) * (1.0f / 512.0f);
    const float d0 = z0 - mean, d1 = z1 - mean;
    const float var = blk_sum(d0 * d0 + d1 * d1) * (1.0f / 512.0f);
    const float inv = rsqrtf(var + 1e-5f);
    float y0 = d0 * inv, y1 = d1 * inv;
    if (g) { y0 = y0 * g[t] + b[t]; y1 = y1 * g[t + 256] + b[t + 256]; }
    Of[o + t] = y0;              Of[o + t + 256] = y1;
    Oh[o + t] = (h16)y0;         Oh[o + t + 256] = (h16)y1;
}

// ------------------------- bias + silu + residual + affine LN --------------
__global__ __launch_bounds__(TPB) void ffn_ep_k(const float* __restrict__ T,
                                                const float* __restrict__ fb,
                                                const float* __restrict__ X,
                                                const float* __restrict__ g,
                                                const float* __restrict__ b,
                                                float* __restrict__ Of,
                                                h16* __restrict__ Oh) {
    const long o = (long)blockIdx.x * 512;
    const int t = threadIdx.x;
    const float u0 = T[o + t]       + fb[t];
    const float u1 = T[o + t + 256] + fb[t + 256];
    const float h0 = u0 / (1.f + __expf(-u0));
    const float h1 = u1 / (1.f + __expf(-u1));
    const float z0 = h0 + X[o + t];
    const float z1 = h1 + X[o + t + 256];
    const float mean = blk_sum(z0 + z1) * (1.0f / 512.0f);
    const float d0 = z0 - mean, d1 = z1 - mean;
    const float var = blk_sum(d0 * d0 + d1 * d1) * (1.0f / 512.0f);
    const float inv = rsqrtf(var + 1e-5f);
    const float y0 = d0 * inv * g[t]       + b[t];
    const float y1 = d1 * inv * g[t + 256] + b[t + 256];
    Of[o + t] = y0;              Of[o + t + 256] = y1;
    Oh[o + t] = (h16)y0;         Oh[o + t + 256] = (h16)y1;
}

// ------------------------- big softmax (in place, 32000 cols) --------------
__global__ __launch_bounds__(TPB) void softmax_big_k(float* __restrict__ S, int cols) {
    float* s = S + (long)blockIdx.x * cols;
    const int t = threadIdx.x;
    float m = -3.0e38f;
    for (int i = t; i < cols; i += TPB) m = fmaxf(m, s[i]);
    m = blk_max(m);
    float sum = 0.f;
    for (int i = t; i < cols; i += TPB) sum += __expf(s[i] - m);
    sum = blk_sum(sum);
    const float inv = 1.0f / sum;
    for (int i = t; i < cols; i += TPB) s[i] = __expf(s[i] - m) * inv;
}

// ---------------------------------------------------------------------------
extern "C" void kernel_launch(void* const* d_in, const int* in_sizes, int n_in,
                              void* d_out, int out_size, void* d_ws, size_t ws_size,
                              hipStream_t stream) {
    (void)in_sizes; (void)n_in; (void)out_size; (void)ws_size;
    const int  Dm = 512, Sq = 2048, Bb = 2, Lc = 3, Vv = 32000;
    const long SD  = (long)Bb * Sq * Dm;
    const long SS  = (long)Bb * Sq * Sq;
    const long DD  = (long)Dm * Dm;
    const long LDD = (long)Lc * DD;
    const long DV  = (long)Dm * Vv;
    const int  BS  = Bb * Sq;                // 4096 rows

    const float* in_x   = (const float*)d_in[0];
    const float* in_y   = (const float*)d_in[1];
    const float* enc_wq = (const float*)d_in[2];
    const float* enc_wk = (const float*)d_in[3];
    const float* enc_wv = (const float*)d_in[4];
    const float* enc_fw = (const float*)d_in[5];
    const float* enc_fb = (const float*)d_in[6];
    const float* enc_g  = (const float*)d_in[7];
    const float* enc_b  = (const float*)d_in[8];
    const float* dec_swq = (const float*)d_in[9];
    const float* dec_swk = (const float*)d_in[10];
    const float* dec_swv = (const float*)d_in[11];
    const float* dec_cwq = (const float*)d_in[12];
    const float* dec_cwk = (const float*)d_in[13];
    const float* dec_cwv = (const float*)d_in[14];
    const float* dec_fw  = (const float*)d_in[15];
    const float* dec_fb  = (const float*)d_in[16];
    const float* dec_g   = (const float*)d_in[17];
    const float* dec_b   = (const float*)d_in[18];
    const float* fc_w    = (const float*)d_in[19];
    float* out = (float*)d_out;

    // ---- workspace layout (256B aligned) ----
    char* w = (char*)d_ws;
    size_t off = 0;
    auto alloc = [&](size_t bytes) -> void* {
        void* p = w + off;
        off = (off + bytes + 255) & ~(size_t)255;
        return p;
    };
    // all weight copies are stored TRANSPOSED: [out_dim][in_dim], k-contiguous
    h16* wqeh = (h16*)alloc(LDD * 2);
    h16* wkeh = (h16*)alloc(LDD * 2);
    h16* wveh = (h16*)alloc(LDD * 2);
    h16* fweh = (h16*)alloc(LDD * 2);
    h16* swqh = (h16*)alloc(LDD * 2);
    h16* swkh = (h16*)alloc(LDD * 2);
    h16* swvh = (h16*)alloc(LDD * 2);
    h16* cwqh = (h16*)alloc(LDD * 2);
    h16* cwkh = (h16*)alloc(LDD * 2);
    h16* cwvh = (h16*)alloc(LDD * 2);
    h16* fwdh = (h16*)alloc(LDD * 2);
    h16* fch  = (h16*)alloc(DV * 2);         // [V][D]
    h16* xh   = (h16*)alloc(SD * 2);
    h16* outh = (h16*)alloc(SD * 2);
    h16* hh   = (h16*)alloc(SD * 2);
    h16* qh   = (h16*)alloc(SD * 2);
    h16* kh   = (h16*)alloc(SD * 2);
    h16* vh   = (h16*)alloc(SD * 2);
    h16* ph   = (h16*)alloc(SS * 2);
    float* xf   = (float*)alloc(SD * 4);
    float* outf = (float*)alloc(SD * 4);
    float* hf   = (float*)alloc(SD * 4);
    float* tf   = (float*)alloc(SD * 4);
    float* sf   = (float*)alloc(SS * 4);

    auto cvt = [&](const float* s, h16* d, long n) {
        long blocks = (n + TPB * 8 - 1) / (TPB * 8);
        if (blocks > 16384) blocks = 16384;
        cvt_k<<<dim3((unsigned)blocks), dim3(TPB), 0, stream>>>(s, d, n);
    };
    auto cvt_t = [&](const float* s, h16* d, int K, int N, int layers) {
        dim3 grid((unsigned)(N / 32), (unsigned)(K / 32), (unsigned)layers);
        cvt_t_k<<<grid, dim3(TPB), 0, stream>>>(s, d, K, N);
    };

    // weights -> f16 transposed; activations -> f16 row-major
    cvt_t(enc_wq, wqeh, Dm, Dm, Lc);  cvt_t(enc_wk, wkeh, Dm, Dm, Lc);
    cvt_t(enc_wv, wveh, Dm, Dm, Lc);  cvt_t(enc_fw, fweh, Dm, Dm, Lc);
    cvt_t(dec_swq, swqh, Dm, Dm, Lc); cvt_t(dec_swk, swkh, Dm, Dm, Lc);
    cvt_t(dec_swv, swvh, Dm, Dm, Lc);
    cvt_t(dec_cwq, cwqh, Dm, Dm, Lc); cvt_t(dec_cwk, cwkh, Dm, Dm, Lc);
    cvt_t(dec_cwv, cwvh, Dm, Dm, Lc);
    cvt_t(dec_fw, fwdh, Dm, Dm, Lc);
    cvt_t(fc_w, fch, Dm, Vv, 1);
    cvt(in_x, xh, SD);  cvt(in_y, outh, SD);
    hipMemcpyAsync(xf,   in_x, SD * sizeof(float), hipMemcpyDeviceToDevice, stream);
    hipMemcpyAsync(outf, in_y, SD * sizeof(float), hipMemcpyDeviceToDevice, stream);

    // bt=1: B is [N,K] (transposed weights / K^T). bt=0: B is [K,N] (P@V).
    auto gemm = [&](const h16* A, const h16* B, float* C, h16* Ch,
                    int M, int N, int K, long sA, long sB, long sC,
                    int batch, float alpha, int bt) {
        dim3 grid((unsigned)(N / 128), (unsigned)(M / 128), (unsigned)batch);
        dim3 blk(TPB);
        if (bt && Ch) gemm_h16<1, 1><<<grid, blk, 0, stream>>>(A, B, nullptr, Ch, M, N, K, sA, sB, sC, alpha);
        else if (bt)  gemm_h16<1, 0><<<grid, blk, 0, stream>>>(A, B, C, nullptr, M, N, K, sA, sB, sC, alpha);
        else          gemm_h16<0, 0><<<grid, blk, 0, stream>>>(A, B, C, nullptr, M, N, K, sA, sB, sC, alpha);
    };

    const float inv_scale = 1.0f / 8.0f;     // 1/sqrt(D/H)
    const long  sQK = (long)Sq * Dm;
    const long  sSS = (long)Sq * Sq;

    // single-head attention: q from qsrc, k/v from ctx; out = LN(P@V + resid)
    auto attn = [&](const h16* qsrc, const h16* ctx,
                    const h16* wq_, const h16* wk_, const h16* wv_,
                    const float* resid, float* of, h16* oh) {
        gemm(qsrc, wq_, nullptr, qh, BS, Dm, Dm, 0, 0, 0, 1, 1.f, 1);
        gemm(ctx,  wk_, nullptr, kh, BS, Dm, Dm, 0, 0, 0, 1, 1.f, 1);
        gemm(ctx,  wv_, nullptr, vh, BS, Dm, Dm, 0, 0, 0, 1, 1.f, 1);
        gemm(qh, kh, sf, nullptr, Sq, Sq, Dm, sQK, sQK, sSS, Bb, inv_scale, 1);  // Q @ K^T
        softmax2048_k<<<dim3((unsigned)BS), dim3(TPB), 0, stream>>>(sf, ph);
        gemm(ph, vh, tf, nullptr, Sq, Dm, Sq, sSS, sQK, sQK, Bb, 1.f, 0);        // P @ V
        add_ln_k<<<dim3((unsigned)BS), dim3(TPB), 0, stream>>>(tf, resid, nullptr, nullptr, of, oh);
    };

    auto ffn = [&](const h16* srch, const float* residf, const h16* fw_,
                   const float* fb_, const float* g_, const float* b_,
                   float* of, h16* oh) {
        gemm(srch, fw_, tf, nullptr, BS, Dm, Dm, 0, 0, 0, 1, 1.f, 1);
        ffn_ep_k<<<dim3((unsigned)BS), dim3(TPB), 0, stream>>>(tf, fb_, residf, g_, b_, of, oh);
    };

    // ---- encoder stacks ----
    for (int i = 0; i < Lc; ++i) {
        attn(xh, xh, wqeh + i * DD, wkeh + i * DD, wveh + i * DD, xf, xf, xh);
        ffn(xh, xf, fweh + i * DD, enc_fb + i * Dm, enc_g + i * Dm, enc_b + i * Dm, xf, xh);
    }

    // ---- decoder stacks (x = encoder output, constant; out = running state) ----
    for (int i = 0; i < Lc; ++i) {
        attn(xh, xh, swqh + i * DD, swkh + i * DD, swvh + i * DD, xf, hf, hh);    // self-attn on enc out
        attn(hh, outh, cwqh + i * DD, cwkh + i * DD, cwvh + i * DD, hf, hf, hh);  // cross: K,V from out
        ffn(hh, hf, fwdh + i * DD, dec_fb + i * Dm, dec_g + i * Dm, dec_b + i * Dm, outf, outh);
    }

    // ---- vocab projection + softmax ----
    gemm(outh, fch, out, nullptr, BS, Vv, Dm, 0, 0, 0, 1, 1.f, 1);
    softmax_big_k<<<dim3((unsigned)BS), dim3(TPB), 0, stream>>>(out, Vv);
}
